// sLSTM_56281251447005
// MI455X (gfx1250) — matrix operand order repaired and checked by reading, not compiled, not run on hardware
//
#include <hip/hip_runtime.h>

// ---------------------------------------------------------------------------
// sLSTM (2 layers) for MI455X / gfx1250, wave32 + WMMA bf16.
// B=16, S=2048, D=512, H=512.
//  - Projection GEMMs: v_wmma_f32_16x16x32_bf16, 128x128 block tiles,
//    register double-buffered staging, both A and B fragments load as
//    ds_load_b128 (B stage stored transposed [col][k]).
//  - Sequential scan: 16 persistent workgroups, W_h slice resident in LDS
//    (128KB/WGP), h exchanged through L2 via global_load_async_to_lds_b128
//    (ASYNCcnt), release/acquire flag sync + s_sleep backoff, dual WMMA
//    accumulators to break the WMMA->WMMA RAW hazard chain.
// ---------------------------------------------------------------------------

typedef unsigned short u16;
typedef unsigned int   u32;
typedef unsigned long long u64;
typedef __attribute__((ext_vector_type(16))) __bf16 v16bf;
typedef __attribute__((ext_vector_type(8)))  float  v8f;

#define BB   16
#define SS   2048
#define PH   2560   /* 5H */

__device__ __forceinline__ u16 f2bf(float f) {
    u32 u = __float_as_uint(f);
    u32 r = (u + 0x7FFFu + ((u >> 16) & 1u)) >> 16;   // round-to-nearest-even
    return (u16)r;
}

// CDNA5 async global->LDS (ASYNCcnt path); no clang builtin on ROCm 7.2.
__device__ __forceinline__ void async_ld_b128(u32 lds_byte_off, u64 gaddr) {
    asm volatile("global_load_async_to_lds_b128 %0, %1, off"
                 :: "v"(lds_byte_off), "v"(gaddr) : "memory");
}
__device__ __forceinline__ void wait_async0() {
    asm volatile("s_wait_asynccnt 0x0" ::: "memory");
}

// ---------------------------------------------------------------------------
// fp32 -> bf16 (optionally column-sliced source); packed u32 stores.
// ---------------------------------------------------------------------------
__global__ __launch_bounds__(256) void k_cvt(const float* __restrict__ src,
                                             u16* __restrict__ dst,
                                             int rows, int cols, int src_stride) {
    long long n2 = (long long)rows * cols / 2;
    for (long long i = (long long)blockIdx.x * 256 + threadIdx.x; i < n2;
         i += (long long)gridDim.x * 256) {
        long long e = i * 2;
        int r = (int)(e / cols), c = (int)(e % cols);
        const float* s = src + (long long)r * src_stride + c;
        u32 pk = (u32)f2bf(s[0]) | ((u32)f2bf(s[1]) << 16);
        *(u32*)&dst[e] = pk;
    }
}

__global__ __launch_bounds__(256) void k_zero(u32* __restrict__ p, long long n) {
    for (long long i = (long long)blockIdx.x * 256 + threadIdx.x; i < n;
         i += (long long)gridDim.x * 256) p[i] = 0u;
}

// ---------------------------------------------------------------------------
// Projection GEMM: C(M x 2048) = A(M x 512 bf16) * W(512 x 2048 bf16) + bias
// Block tile 128x128, 8 waves (4 along M, 2 along N), wave tile 32x64.
// Register double-buffer: loads for stage kb+1 issue before wmma of stage kb.
// ---------------------------------------------------------------------------
__global__ __launch_bounds__(256) void k_proj(const u16* __restrict__ A,
                                              const u16* __restrict__ W,
                                              const float* __restrict__ bias,
                                              float* __restrict__ C) {
    __shared__ u16 Al[128 * 32];   // [row][k]
    __shared__ u16 Bl[128 * 32];   // [col][k]  (transposed -> b128 frag loads)
    const int tid  = threadIdx.x;
    const int lane = tid & 31, wid = tid >> 5;
    const int row0 = blockIdx.x * 128, col0 = blockIdx.y * 128;
    const int wm = (wid & 3) * 32;    // wave row offset in block tile
    const int wn = (wid >> 2) * 64;   // wave col offset in block tile

    union Acc { v8f v; float f[8]; } acc[2][4];
    for (int i = 0; i < 2; i++)
        for (int j = 0; j < 4; j++)
            for (int k = 0; k < 8; k++) acc[i][j].f[k] = 0.f;

    const int khalf = (lane >> 4) * 8;    // A-fragment half-lane K offset
    const int kh16  = (lane >> 4) * 16;   // unused; B fragments use khalf too
    const int arow  = lane & 15;

    u32 aR[8], bR[8];
    // stage-load lambda: coalesced u32 global reads into registers
    auto load_stage = [&](int kb) {
#pragma unroll
        for (int b = 0; b < 8; b++) {
            int i = b * 256 + tid;
            int r = i >> 4, c = (i & 15) * 2;
            aR[b] = *(const u32*)&A[(long long)(row0 + r) * 512 + kb * 32 + c];
        }
#pragma unroll
        for (int b = 0; b < 8; b++) {
            int i = b * 256 + tid;
            int k = i >> 6, c = (i & 63) * 2;
            bR[b] = *(const u32*)&W[(long long)(kb * 32 + k) * 2048 + col0 + c];
        }
    };

    load_stage(0);
    for (int kb = 0; kb < 16; ++kb) {
        __syncthreads();
        // commit staged registers to LDS (B transposed)
#pragma unroll
        for (int b = 0; b < 8; b++) {
            int i = b * 256 + tid;
            int r = i >> 4, c = (i & 15) * 2;
            *(u32*)&Al[r * 32 + c] = aR[b];
        }
#pragma unroll
        for (int b = 0; b < 8; b++) {
            int i = b * 256 + tid;
            int k = i >> 6, c = (i & 63) * 2;
            Bl[c * 32 + k]       = (u16)(bR[b] & 0xffffu);
            Bl[(c + 1) * 32 + k] = (u16)(bR[b] >> 16);
        }
        __syncthreads();
        if (kb + 1 < 16) load_stage(kb + 1);   // overlap global latency w/ wmma

        union Fr { v16bf v; u32 u[8]; } afr[2], bfr[4];
        // A fragments: ISA 16-bit A 16x32 layout; contiguous pairs -> b128
#pragma unroll
        for (int i = 0; i < 2; i++) {
            int r = wm + i * 16 + arow;
#pragma unroll
            for (int p = 0; p < 8; p++) {
                int K = ((p >= 4) ? 16 : 0) + khalf + 2 * (p & 3);
                afr[i].u[p] = *(const u32*)&Al[r * 32 + K];
            }
        }
        // B fragments: same K layout as A (ISA 16-bit 16x32 order); [col][k]
#pragma unroll
        for (int j = 0; j < 4; j++) {
            int c = wn + j * 16 + (lane & 15);
#pragma unroll
            for (int p = 0; p < 8; p++)
                bfr[j].u[p] = *(const u32*)&Bl[c * 32 + ((p >= 4) ? 16 : 0) + khalf + 2 * (p & 3)];
        }
#pragma unroll
        for (int i = 0; i < 2; i++)
#pragma unroll
            for (int j = 0; j < 4; j++)
                acc[i][j].v = __builtin_amdgcn_wmma_f32_16x16x32_bf16(
                    false, afr[i].v, false, bfr[j].v, (short)0, acc[i][j].v,
                    false, false);
    }

    // Store: C fragment layout (VGPR v -> M=v (lanes 0-15) / M=8+v (16-31))
    const int moff = (lane >> 4) << 3;
    const int nlan = lane & 15;
    for (int i = 0; i < 2; i++)
        for (int j = 0; j < 4; j++) {
            int gc = col0 + wn + j * 16 + nlan;
            float bv = bias[gc];
            for (int v = 0; v < 8; v++) {
                int gr = row0 + wm + i * 16 + moff + v;
                C[(long long)gr * 2048 + gc] = acc[i][j].f[v] + bv;
            }
        }
}

// ---------------------------------------------------------------------------
// Persistent scan: 16 blocks x 256 threads. Block b owns h-indices
// [b*32, b*32+32) => 128 W_h columns (4 gates x 32) resident in LDS.
// ---------------------------------------------------------------------------
__global__ __launch_bounds__(256) void k_scan(const u16*  __restrict__ Wh,    // 512 x 2048 bf16
                                              const float* __restrict__ ip,   // (B*S) x 2048 f32
                                              const float* __restrict__ bh,   // 2048 f32
                                              u16*  __restrict__ hbuf,        // 2 x 16 x 512 bf16
                                              float* __restrict__ out,        // (B*S) x 512 f32
                                              int*  __restrict__ flags) {     // S ints
    extern __shared__ char smem[];
    u16*   Wl = (u16*)smem;                         // [128 col][512 k]  128KB
    u16*   Hl = (u16*)(smem + 131072);              // [16 b][512 k]      16KB
    float* HP = (float*)(smem + 131072 + 16384);    // [16 b][128 col]     8KB
    const u32 hl_off = (u32)((const char*)Hl - (const char*)smem);

    const int tid = threadIdx.x, lane = tid & 31, wid = tid >> 5;
    const int j0 = blockIdx.x * 32;

    // One-time weight slice load (register-batched; coalesced u32 reads,
    // transposed u16 LDS writes): Wl[c*512+k] = Wh[k, gate*512 + j0 + j]
    for (int it = 0; it < 16; ++it) {
        u32 tmp[8];
#pragma unroll
        for (int b = 0; b < 8; b++) {
            int idx = it * 2048 + b * 256 + tid;
            int k = idx >> 6, c = (idx & 63) * 2;
            int g = c >> 5, j = c & 31;
            tmp[b] = *(const u32*)&Wh[(long long)k * 2048 + g * 512 + j0 + j];
        }
#pragma unroll
        for (int b = 0; b < 8; b++) {
            int idx = it * 2048 + b * 256 + tid;
            int k = idx >> 6, c = (idx & 63) * 2;
            Wl[c * 512 + k]       = (u16)(tmp[b] & 0xffffu);
            Wl[(c + 1) * 512 + k] = (u16)(tmp[b] >> 16);
        }
    }

    float creg[2] = {0.f, 0.f}, mreg[2] = {0.f, 0.f};
    const int khalf = (lane >> 4) * 8;
    const int kh16  = (lane >> 4) * 16;
    const int arow  = lane & 15;
    const int bcol  = wid * 16 + (lane & 15);
    __syncthreads();

    for (int t = 0; t < SS; ++t) {
        if (t > 0) {
            // free HW barrier if dispatched as a cluster; ISA NOP otherwise
            if (__builtin_amdgcn_cluster_id_x() > 0)
                __builtin_amdgcn_s_cluster_barrier();
            if (tid == 0) {
                while (__hip_atomic_load(&flags[t - 1], __ATOMIC_ACQUIRE,
                                         __HIP_MEMORY_SCOPE_AGENT) < 16)
                    __builtin_amdgcn_s_sleep(2);
            }
            __syncthreads();
        }
        // Async-gather h(t-1) (16KB bf16) straight into LDS, no VGPR staging
        {
            const char* hsrc = (const char*)(hbuf + (((t & 1) ^ 1) * (16 * 512)));
#pragma unroll
            for (int q = 0; q < 4; q++) {
                u32 boff = (u32)(tid + q * 256) * 16u;
                async_ld_b128(hl_off + boff, (u64)(size_t)(hsrc + boff));
            }
            wait_async0();
        }
        __syncthreads();

        // Recurrent GEMM: wave `wid` computes the 16x16 tile of cols [wid*16,+16)
        // Dual accumulators break the WMMA->WMMA RAW chain (ISA 7.12.1).
        union Acc { v8f v; float f[8]; } acc0, acc1;
        for (int v = 0; v < 8; v++) { acc0.f[v] = 0.f; acc1.f[v] = 0.f; }
#pragma unroll
        for (int kb = 0; kb < 16; kb += 2) {
            union Fr { v16bf v; u32 u[8]; } af0, bf0, af1, bf1;
#pragma unroll
            for (int p = 0; p < 8; p++) {
                int Ka = ((p >= 4) ? 16 : 0) + khalf + 2 * (p & 3);
                af0.u[p] = *(const u32*)&Hl[arow * 512 + kb * 32 + Ka];
                bf0.u[p] = *(const u32*)&Wl[bcol * 512 + kb * 32 + Ka];
                af1.u[p] = *(const u32*)&Hl[arow * 512 + (kb + 1) * 32 + Ka];
                bf1.u[p] = *(const u32*)&Wl[bcol * 512 + (kb + 1) * 32 + Ka];
            }
            acc0.v = __builtin_amdgcn_wmma_f32_16x16x32_bf16(
                false, af0.v, false, bf0.v, (short)0, acc0.v, false, false);
            acc1.v = __builtin_amdgcn_wmma_f32_16x16x32_bf16(
                false, af1.v, false, bf1.v, (short)0, acc1.v, false, false);
        }
        {
            int n = lane & 15, moff = (lane >> 4) << 3;
            for (int v = 0; v < 8; v++)
                HP[(moff + v) * 128 + wid * 16 + n] = acc0.f[v] + acc1.f[v];
        }
        __syncthreads();

        // Gating: thread owns (b,j) pairs q = 2*tid, 2*tid+1; c/m live in regs
        for (int s = 0; s < 2; s++) {
            int q = tid * 2 + s;
            int b = q >> 5, j = q & 31;
            const float* ipb = ip + (long long)(b * SS + t) * 2048;
            float ih = HP[b * 128 +   0 + j] + bh[        j0 + j];
            float fh = HP[b * 128 +  32 + j] + bh[ 512 + j0 + j];
            float zh = HP[b * 128 +  64 + j] + bh[1024 + j0 + j];
            float oh = HP[b * 128 +  96 + j] + bh[1536 + j0 + j];
            float ix = ipb[        j0 + j];
            float fx = ipb[ 512 + j0 + j];
            float zx = ipb[1024 + j0 + j];
            float ox = ipb[1536 + j0 + j];

            float log_i = fminf(fmaxf(ix + ih, -10.f), 10.f);
            float log_f = fminf(fmaxf(fx + fh, -10.f), 10.f);
            float m_new = fmaxf(log_f + mreg[s], log_i);
            float i_t = __expf(log_i - m_new);
            float f_t = __expf(log_f + mreg[s] - m_new);
            float z_t = tanhf(zx + zh);
            float c_new = f_t * creg[s] + i_t * z_t;
            float o_t = 1.f / (1.f + __expf(-(ox + oh)));
            float h_new = o_t * tanhf(c_new);
            creg[s] = c_new; mreg[s] = m_new;

            out[(long long)(b * SS + t) * 512 + j0 + j] = h_new;
            hbuf[(t & 1) * (16 * 512) + b * 512 + j0 + j] = f2bf(h_new);
            if (t + 1 < SS) __builtin_prefetch(&ipb[2048 + j0 + j], 0, 0);
        }
        __threadfence();   // agent-scope release of this block's h(t)/out writes
        __syncthreads();
        if (tid == 0)
            __hip_atomic_fetch_add(&flags[t], 1, __ATOMIC_RELEASE,
                                   __HIP_MEMORY_SCOPE_AGENT);
    }
}

// ---------------------------------------------------------------------------
// LayerNorm over H=512; one block per (b,t) row. Writes fp32 and/or bf16.
// ---------------------------------------------------------------------------
__global__ __launch_bounds__(256) void k_ln(const float* __restrict__ in,
                                            const float* __restrict__ gamma,
                                            const float* __restrict__ beta,
                                            float* __restrict__ outF,
                                            u16*  __restrict__ outBF) {
    __shared__ float rs[8], rss[8];
    const long long row = blockIdx.x;
    const float* r = in + row * 512;
    const int tid = threadIdx.x;
    float v[2], s = 0.f, ss = 0.f;
    for (int i = 0; i < 2; i++) {
        v[i] = r[tid + i * 256];
        s += v[i]; ss += v[i] * v[i];
    }
    for (int off = 16; off > 0; off >>= 1) {
        s  += __shfl_down(s,  off, 32);
        ss += __shfl_down(ss, off, 32);
    }
    if ((tid & 31) == 0) { rs[tid >> 5] = s; rss[tid >> 5] = ss; }
    __syncthreads();
    float ts = 0.f, tss = 0.f;
    for (int w = 0; w < 8; w++) { ts += rs[w]; tss += rss[w]; }
    float mu  = ts * (1.f / 512.f);
    float var = tss * (1.f / 512.f) - mu * mu;
    float rstd = rsqrtf(var + 1e-5f);
    for (int i = 0; i < 2; i++) {
        int c = tid + i * 256;
        float y = (v[i] - mu) * rstd * gamma[c] + beta[c];
        if (outF)  outF[row * 512 + c]  = y;
        if (outBF) outBF[row * 512 + c] = f2bf(y);
    }
}

// ---------------------------------------------------------------------------
extern "C" void kernel_launch(void* const* d_in, const int* in_sizes, int n_in,
                              void* d_out, int out_size, void* d_ws, size_t ws_size,
                              hipStream_t stream) {
    (void)in_sizes; (void)n_in; (void)out_size; (void)ws_size;

    const float* x     = (const float*)d_in[0];
    const float* W_in0 = (const float*)d_in[1];
    const float* b_in0 = (const float*)d_in[2];
    const float* W_h0  = (const float*)d_in[3];
    const float* b_h0  = (const float*)d_in[4];
    const float* g0    = (const float*)d_in[5];
    const float* be0   = (const float*)d_in[6];
    const float* W_in1 = (const float*)d_in[7];
    const float* b_in1 = (const float*)d_in[8];
    const float* W_h1  = (const float*)d_in[9];
    const float* b_h1  = (const float*)d_in[10];
    const float* g1    = (const float*)d_in[11];
    const float* be1   = (const float*)d_in[12];
    float* outF = (float*)d_out;

    // Workspace carve-up (~356 MB)
    char* ws = (char*)d_ws;
    size_t off = 0;
    u16* xbf  = (u16*)(ws + off); off += (size_t)BB * SS * 512 * 2;    // 32 MB
    u16* wibf = (u16*)(ws + off); off += (size_t)512 * 2048 * 2;       //  2 MB
    u16* whbf = (u16*)(ws + off); off += (size_t)512 * 2048 * 2;       //  2 MB
    float* ip = (float*)(ws + off); off += (size_t)BB * SS * 2048 * 4; // 256 MB
    float* o0 = (float*)(ws + off); off += (size_t)BB * SS * 512 * 4;  //  64 MB
    u16* hbuf = (u16*)(ws + off); off += (size_t)2 * 16 * 512 * 2;
    int* flags = (int*)(ws + off); off += (size_t)SS * 4;

    const dim3 projGrid(BB * SS / 128, 2048 / 128);
    const size_t scanLds = 131072 + 16384 + 8192;   // 152 KB (fits 320 KB WGP LDS)

    // ---- Layer 0 ----
    k_cvt<<<2048, 256, 0, stream>>>(x,     xbf,  BB * SS, 512, 512);
    k_cvt<<<1024, 256, 0, stream>>>(W_in0, wibf, 512, 2048, PH);
    k_cvt<<<1024, 256, 0, stream>>>(W_h0,  whbf, 512, 2048, 2048);
    k_proj<<<projGrid, 256, 0, stream>>>(xbf, wibf, b_in0, ip);
    k_zero<<<32, 256, 0, stream>>>((u32*)hbuf, 2 * 16 * 512 / 2);
    k_zero<<<8, 256, 0, stream>>>((u32*)flags, SS);
    k_scan<<<16, 256, scanLds, stream>>>(whbf, ip, b_h0, hbuf, o0, flags);
    k_ln<<<BB * SS, 256, 0, stream>>>(o0, g0, be0, nullptr, xbf);  // LN0 -> bf16 layer-1 input

    // ---- Layer 1 ----
    k_cvt<<<1024, 256, 0, stream>>>(W_in1, wibf, 512, 2048, PH);
    k_cvt<<<1024, 256, 0, stream>>>(W_h1,  whbf, 512, 2048, 2048);
    k_proj<<<projGrid, 256, 0, stream>>>(xbf, wibf, b_in1, ip);
    k_zero<<<32, 256, 0, stream>>>((u32*)hbuf, 2 * 16 * 512 / 2);
    k_zero<<<8, 256, 0, stream>>>((u32*)flags, SS);
    k_scan<<<16, 256, scanLds, stream>>>(whbf, ip, b_h1, hbuf, outF, flags);
    k_ln<<<BB * SS, 256, 0, stream>>>(outF, g1, be1, outF, nullptr);  // in-place final LN
}
